// KMeans_54133767799018
// MI455X (gfx1250) — matrix operand order repaired and checked by reading, not compiled, not run on hardware
//
#include <hip/hip_runtime.h>
#include <hip/hip_bf16.h>

typedef __attribute__((ext_vector_type(16))) __bf16 v16bf;
typedef __attribute__((ext_vector_type(8)))  __bf16 v8bf;
typedef __attribute__((ext_vector_type(8)))  float  v8f;
typedef __attribute__((ext_vector_type(4)))  int    v4i;

#define DDIM 128
#define TILE_ROWS 128     // rows of x per block (8 waves x 16 rows)
#define XCH 136           // padded f32 stride for x rows in LDS (544B -> 8-bank skew)
#define CCH 144           // padded bf16 stride for center rows in LDS (288B -> 8-bank skew)
#define CCHUNK 128        // centers per LDS buffer (double-buffered)
#define CBUF_BYTES (CCHUNK * CCH * 2)                       // 36864 per buffer
#define SMEM_BYTES (2 * CBUF_BYTES + TILE_ROWS * XCH * 4 + TILE_ROWS * 4) // 73728+69632+512

// ---- async global->LDS copy (CDNA5 GLOBAL_LOAD_ASYNC_TO_LDS_B128), with fallback ----
#if __has_builtin(__builtin_amdgcn_global_load_async_to_lds_b128) && \
    __has_builtin(__builtin_amdgcn_s_wait_asynccnt)
#define HAVE_ASYNC 1
#else
#define HAVE_ASYNC 0
#endif

static __device__ __forceinline__ void copy16(const void* g, void* l) {
#if HAVE_ASYNC
    __builtin_amdgcn_global_load_async_to_lds_b128(
        (__attribute__((address_space(1))) v4i*)(g),
        (__attribute__((address_space(3))) v4i*)(l), 0, 0);
#else
    *(uint4*)l = *(const uint4*)g;
#endif
}
static __device__ __forceinline__ void copy_wait() {
#if HAVE_ASYNC
    __builtin_amdgcn_s_wait_asynccnt(0);
#endif
}

// ---------------- prep: centers f32 -> bf16 in ws, plus ||c||^2 ----------------
__global__ __launch_bounds__(128)
void kmeans_prep(const float* __restrict__ centers, __bf16* __restrict__ cbf,
                 float* __restrict__ c2) {
    __shared__ float partial[4];
    const int c = blockIdx.x, t = threadIdx.x;
    float v = centers[c * DDIM + t];
    cbf[c * DDIM + t] = (__bf16)v;
    float s = v * v;
    #pragma unroll
    for (int off = 16; off >= 1; off >>= 1) s += __shfl_down(s, off, 32);
    if ((t & 31) == 0) partial[t >> 5] = s;
    __syncthreads();
    if (t == 0) c2[c] = partial[0] + partial[1] + partial[2] + partial[3];
}

// ---------------- zero: clear sums region of d_out + counts in ws ----------------
__global__ void kmeans_zero(float* __restrict__ sums, float* __restrict__ counts,
                            int nsums, int ncounts) {
    int i = blockIdx.x * blockDim.x + threadIdx.x;
    if (i < nsums) sums[i] = 0.0f;
    else if (i < nsums + ncounts) counts[i - nsums] = 0.0f;
}

// ---------------- assign: WMMA distance ranking + argmin + segment sums ----------------
__global__ __launch_bounds__(256)
void kmeans_assign(const float* __restrict__ x, const __bf16* __restrict__ cbf,
                   const float* __restrict__ c2, float* __restrict__ outIdx,
                   float* __restrict__ sums, float* __restrict__ counts, int K) {
    extern __shared__ char smem[];
    float* sX  = (float*)(smem + 2 * CBUF_BYTES);
    int*   sIx = (int*)(smem + 2 * CBUF_BYTES + TILE_ROWS * XCH * 4);

    const int tid  = threadIdx.x;
    const int lane = tid & 31;
    const int wave = tid >> 5;
    const int rowBase = blockIdx.x * TILE_ROWS;
    const int nchunks = K / CCHUNK;

    // ---- prologue: async-stage x tile (f32, padded) + center chunk 0 ----
    {
        const uint4* xg = (const uint4*)(x + (size_t)rowBase * DDIM);
        for (int i = tid; i < TILE_ROWS * (DDIM / 4); i += 256) {
            int r = i >> 5, c4 = i & 31;                  // 32 uint4 per x row
            copy16(xg + (size_t)r * 32 + c4, (char*)(sX + r * XCH) + c4 * 16);
        }
        const uint4* cg = (const uint4*)cbf;              // chunk 0 starts at center 0
        for (int i = tid; i < CCHUNK * (DDIM / 8); i += 256) {
            int r = i >> 4, c16 = i & 15;                 // 16 uint4 per center row
            copy16(cg + i, smem + r * (CCH * 2) + c16 * 16);
        }
    }
    copy_wait();
    __syncthreads();

    // ---- build 4 A-fragments (16x32 bf16) in registers, reused for all tiles ----
    // ISA 16-bit A layout: lane m=L%16, h=L/16; elems 0..7 = K 8h..8h+7, 8..15 = K 16+8h..
    const int m = lane & 15, half = lane >> 4;
    v16bf afrag[4];
    #pragma unroll
    for (int c = 0; c < 4; ++c) {
        const float* row = sX + (wave * 16 + m) * XCH + c * 32 + half * 8;
        v16bf a;
        #pragma unroll
        for (int i = 0; i < 8; ++i) a[i] = (__bf16)row[i];
        #pragma unroll
        for (int i = 0; i < 8; ++i) a[8 + i] = (__bf16)row[16 + i];
        afrag[c] = a;
    }

    float minv[8]; int mini[8];
    #pragma unroll
    for (int r = 0; r < 8; ++r) { minv[r] = 3.4e38f; mini[r] = 0; }

    // ---- main loop: double-buffered center chunks, async prefetch of chunk+1 ----
    int cur = 0;
    for (int chunk = 0; chunk < nchunks; ++chunk) {
        if (chunk + 1 < nchunks) {   // prefetch next chunk into the spare buffer
            const uint4* cg = (const uint4*)(cbf + (size_t)(chunk + 1) * CCHUNK * DDIM);
            char* dst = smem + (size_t)(cur ^ 1) * CBUF_BYTES;
            for (int i = tid; i < CCHUNK * (DDIM / 8); i += 256) {
                int r = i >> 4, c16 = i & 15;
                copy16(cg + i, dst + r * (CCH * 2) + c16 * 16);
            }
        }

        const __bf16* buf = (const __bf16*)(smem + (size_t)cur * CBUF_BYTES);
        for (int t = 0; t < CCHUNK / 16; ++t) {
            // Preload all 4 B fragments first so the 8 ds_load_b128 pipeline.
            // ISA 16-bit B (32x16) layout: col n=L%16, h=L/16; elems 0..7 = K 8h..8h+7, 8..15 = K 16+8h..
            v16bf b[4];
            #pragma unroll
            for (int c = 0; c < 4; ++c) {
                const __bf16* bp = buf + (t * 16 + m) * CCH + c * 32 + half * 8;
                v8bf lo = *(const v8bf*)(bp);
                v8bf hi = *(const v8bf*)(bp + 16);
                v16bf bb;
                #pragma unroll
                for (int i = 0; i < 8; ++i) { bb[i] = lo[i]; bb[8 + i] = hi[i]; }
                b[c] = bb;
            }
            v8f acc = {};
            #pragma unroll
            for (int c = 0; c < 4; ++c)
                acc = __builtin_amdgcn_wmma_f32_16x16x32_bf16(
                        false, afrag[c], false, b[c], (short)0, acc, false, false);

            const int g = chunk * CCHUNK + t * 16 + m;   // this lane's center column
            const float c2v = c2[g];
            #pragma unroll
            for (int r = 0; r < 8; ++r) {                // rank by c2 - 2*dot
                float score = c2v - 2.0f * acc[r];
                if (score < minv[r]) { minv[r] = score; mini[r] = g; }
            }
        }

        copy_wait();        // my async loads for chunk+1 landed
        __syncthreads();    // everyone done reading buf + everyone's loads landed
        cur ^= 1;
    }

    // ---- per-row argmin: butterfly across the 16 lanes sharing a row ----
    #pragma unroll
    for (int r = 0; r < 8; ++r) {
        float v = minv[r]; int idx = mini[r];
        #pragma unroll
        for (int off = 8; off >= 1; off >>= 1) {
            float ov = __shfl_xor(v, off, 16);
            int   oi = __shfl_xor(idx, off, 16);
            if (ov < v || (ov == v && oi < idx)) { v = ov; idx = oi; }
        }
        minv[r] = v; mini[r] = idx;
    }
    if (m == 0) {   // lanes 0 and 16: rows r and 8+r of this wave's 16-row slab
        #pragma unroll
        for (int r = 0; r < 8; ++r) {
            int row = wave * 16 + half * 8 + r;
            outIdx[rowBase + row] = (float)mini[r];
            sIx[row] = mini[r];
        }
    }
    __syncthreads();

    // ---- segment sums from the f32 LDS tile (full precision) ----
    for (int r = wave; r < TILE_ROWS; r += 8) {
        const int cIdx = sIx[r];
        float* dst = sums + (size_t)cIdx * DDIM;
        const float* src = sX + r * XCH;
        for (int d = lane; d < DDIM; d += 32) atomicAdd(dst + d, src[d]);
        if (lane == 0) atomicAdd(counts + cIdx, 1.0f);
    }
}

// ---------------- finalize: new_centers + new_accum ----------------
__global__ __launch_bounds__(256)
void kmeans_final(float* __restrict__ sums /* in: segment sums, out: new_centers */,
                  const float* __restrict__ counts, const float* __restrict__ centers,
                  const float* __restrict__ accum, float* __restrict__ outAccum, int K) {
    int i = blockIdx.x * blockDim.x + threadIdx.x;
    if (i < K * DDIM) {
        int k = i >> 7;  // DDIM == 128
        float cnt = counts[k];
        float lr = 0.9f / accum[k] + 0.1f;
        float s = sums[i];
        float grad = (cnt > 0.0f) ? (s / cnt) : 0.0f;
        sums[i] = lr * grad + (1.0f - lr) * centers[i];
        if (i < K) outAccum[i] = accum[i] + cnt;
    }
}

extern "C" void kernel_launch(void* const* d_in, const int* in_sizes, int n_in,
                              void* d_out, int out_size, void* d_ws, size_t ws_size,
                              hipStream_t stream) {
    const float* x       = (const float*)d_in[0];
    const float* centers = (const float*)d_in[1];
    const float* accum   = (const float*)d_in[2];
    const int N = in_sizes[0] / DDIM;      // 262144
    const int K = in_sizes[2];             // 1024

    float* out      = (float*)d_out;
    float* outIdx   = out;                                  // [N] argmin as float
    float* sums     = out + (size_t)N;                      // [K*D] sums -> new_centers
    float* outAccum = out + (size_t)N + (size_t)K * DDIM;   // [K]

    char* ws = (char*)d_ws;
    __bf16* cbf = (__bf16*)ws;                                           // K*D bf16
    float* c2   = (float*)(ws + (size_t)K * DDIM * 2);                   // K f32
    float* cnts = (float*)(ws + (size_t)K * DDIM * 2 + (size_t)K * 4);   // K f32

    (void)hipFuncSetAttribute((const void*)kmeans_assign,
                              hipFuncAttributeMaxDynamicSharedMemorySize, SMEM_BYTES);

    int nz = K * DDIM + K;
    kmeans_zero<<<(nz + 255) / 256, 256, 0, stream>>>(sums, cnts, K * DDIM, K);
    kmeans_prep<<<K, DDIM, 0, stream>>>(centers, cbf, c2);
    kmeans_assign<<<N / TILE_ROWS, 256, SMEM_BYTES, stream>>>(x, cbf, c2, outIdx, sums, cnts, K);
    kmeans_final<<<(K * DDIM + 255) / 256, 256, 0, stream>>>(sums, cnts, centers, accum, outAccum, K);
}
